// Attention_68289980006461
// MI455X (gfx1250) — matrix operand order repaired and checked by reading, not compiled, not run on hardware
//
#include <hip/hip_runtime.h>

typedef __bf16 bf16_t;
typedef __attribute__((ext_vector_type(16))) __bf16 v16bf;
typedef __attribute__((ext_vector_type(8)))  __bf16 v8bf;
typedef __attribute__((ext_vector_type(4)))  __bf16 v4bf;
typedef __attribute__((ext_vector_type(8)))  float   v8f;
typedef __attribute__((ext_vector_type(4)))  float   v4f;

union Frag { v16bf v; v8bf h[2]; };

// Problem constants
#define NTOK   64      // tokens per window (N)
#define CDIM   256     // channels (C)
#define HEADS  8
#define HD     32      // head dim
#define QKV_N  768     // 3*C

// Padded LDS row strides (elements). Byte stride is an odd multiple of 16 so the 16 lanes of a
// half-wave land on distinct 4-bank quads for ds_load_b128 (conflict-free), while
// staying 16B aligned.
#define XB_STRIDE 264   // xb / aout rows (256 cols)
#define QK_STRIDE 520   // qk rows (512 cols)
#define VT_STRIDE 72    // vT rows (64 tokens)
#define PB_STRIDE 72    // P slab rows (64 cols)

// ws layout: packed bf16 weights in WMMA B-fragment order
static constexpr int WQKV_ELEMS  = 48 * 8 * 32 * 16;  // 196608 (ntile, ktile, lane, 16 elems)
static constexpr int WPROJ_ELEMS = 16 * 8 * 32 * 16;  //  65536

// LDS layout (dynamic shared), bytes:
//  xb   : 64*264 bf16    = 33792   @ 0
//  qk   : 64*520 bf16    = 66560   @ 33792
//  vT   : 256*72 bf16    = 36864   @ 100352
//  mk   : 64*64  f32     = 16384   @ 137216
//  pbuf : 8*16*72 bf16   = 18432   @ 153600
//  aout : 64*264 bf16    = 33792   @ 172032
static constexpr int LDS_BYTES = 205824;

__global__ void pack_weights_kernel(const float* __restrict__ w_qkv,
                                    const float* __restrict__ w_proj,
                                    bf16_t* __restrict__ wqkv_pk,
                                    bf16_t* __restrict__ wproj_pk) {
    int idx = blockIdx.x * 256 + threadIdx.x;
    if (idx < WQKV_ELEMS) {
        int tile = idx >> 9;          // ntile*8 + ktile
        int rem  = idx & 511;         // lane*16 + j
        int lane = rem >> 4;
        int j    = rem & 15;
        int nt = tile >> 3, kt = tile & 7;
        int g  = lane >> 4;
        int ncol = nt * 16 + (lane & 15);
        int k    = kt * 32 + g * 8 + (j & 7) + (j >> 3) * 16;   // B-frag: K = k0 + g*8 + (j&7) + (j>>3)*16
        wqkv_pk[idx] = (bf16_t)w_qkv[k * QKV_N + ncol];
    }
    if (idx < WPROJ_ELEMS) {
        int tile = idx >> 9;
        int rem  = idx & 511;
        int lane = rem >> 4;
        int j    = rem & 15;
        int nt = tile >> 3, kt = tile & 7;
        int g  = lane >> 4;
        int ncol = nt * 16 + (lane & 15);
        int k    = kt * 32 + g * 8 + (j & 7) + (j >> 3) * 16;
        wproj_pk[idx] = (bf16_t)w_proj[k * CDIM + ncol];
    }
}

__launch_bounds__(256)
__global__ void window_attn_kernel(const float* __restrict__ x,
                                   const float* __restrict__ mask,
                                   const float* __restrict__ b_qkv,
                                   const float* __restrict__ b_proj,
                                   const bf16_t* __restrict__ wqkv_pk,
                                   const bf16_t* __restrict__ wproj_pk,
                                   float* __restrict__ out) {
    extern __shared__ char smem[];
    bf16_t* xb   = (bf16_t*)(smem);
    bf16_t* qk   = (bf16_t*)(smem + 33792);
    bf16_t* vT   = (bf16_t*)(smem + 100352);
    float*  mk   = (float*)(smem + 137216);
    bf16_t* pbuf = (bf16_t*)(smem + 153600);
    bf16_t* aout = (bf16_t*)(smem + 172032);

    const int b    = blockIdx.x;
    const int tid  = threadIdx.x;
    const int lane = tid & 31;
    const int wv   = tid >> 5;
    const int ln16 = lane & 15;
    const int g    = lane >> 4;

    // ---------------- Phase 0: stage x (bf16, padded rows) and mask window (f32)
    const float* xg = x + (size_t)b * (NTOK * CDIM);
    #pragma unroll
    for (int i = 0; i < 16; ++i) {
        int idx = i * 1024 + tid * 4;
        int row = idx >> 8, col = idx & 255;
        v4f val = *(const v4f*)(xg + idx);
        v4bf hv;
        hv.x = (bf16_t)val.x; hv.y = (bf16_t)val.y;
        hv.z = (bf16_t)val.z; hv.w = (bf16_t)val.w;
        *(v4bf*)(xb + row * XB_STRIDE + col) = hv;
    }
    const float* mg = mask + (size_t)(b & 63) * (NTOK * NTOK);
    #pragma unroll
    for (int i = 0; i < 4; ++i) {
        int idx = i * 1024 + tid * 4;
        *(v4f*)(mk + idx) = *(const v4f*)(mg + idx);
    }
    __syncthreads();

    // ---------------- Phase 1: QKV = x @ Wqkv + b  (64x256 @ 256x768)
    // Each wave owns 6 n-tiles across ALL 4 m-tiles: B-fragment loaded once from
    // L2, reused by 4 independent WMMA accumulation chains.
    {
        for (int j = 0; j < 6; ++j) {
            const int nt = wv + 8 * j;
            v8f acc[4] = {{}, {}, {}, {}};
            const bf16_t* wp = wqkv_pk + nt * (8 * 512) + lane * 16;
            const bf16_t* ap0 = xb + ln16 * XB_STRIDE + g * 8;
            #pragma unroll
            for (int kt = 0; kt < 8; ++kt) {
                Frag bb;
                bb.h[0] = *(const v8bf*)(wp + kt * 512);
                bb.h[1] = *(const v8bf*)(wp + kt * 512 + 8);
                #pragma unroll
                for (int mt = 0; mt < 4; ++mt) {
                    Frag a;
                    const bf16_t* ap = ap0 + mt * 16 * XB_STRIDE + kt * 32;
                    a.h[0] = *(const v8bf*)(ap);
                    a.h[1] = *(const v8bf*)(ap + 16);
                    acc[mt] = __builtin_amdgcn_wmma_f32_16x16x32_bf16(
                        false, a.v, false, bb.v, (short)0, acc[mt], false, false);
                }
            }
            const int n0  = nt * 16;
            const int col = n0 + ln16;
            const float bias = b_qkv[col];
            if (n0 < 512) {                      // q (scaled) and k -> token-major
                const float scl = (n0 < 256) ? 0.17677669529663687f : 1.0f; // hd^-0.5
                bf16_t* dst = qk + col;
                #pragma unroll
                for (int mt = 0; mt < 4; ++mt)
                    #pragma unroll
                    for (int r = 0; r < 8; ++r)
                        dst[(mt * 16 + r + 8 * g) * QK_STRIDE] =
                            (bf16_t)((acc[mt][r] + bias) * scl);
            } else {                             // v -> transposed (feat-major, padded)
                bf16_t* dst = vT + (col - 512) * VT_STRIDE + 8 * g;
                #pragma unroll
                for (int mt = 0; mt < 4; ++mt)
                    #pragma unroll
                    for (int r = 0; r < 8; ++r)
                        dst[mt * 16 + r] = (bf16_t)(acc[mt][r] + bias);
            }
        }
    }
    __syncthreads();

    // ---------------- Phase 2: per-head attention (wave w = head w)
    {
        const int head = wv;
        bf16_t* myP = pbuf + wv * (16 * PB_STRIDE);
        #pragma unroll 1
        for (int mt = 0; mt < 4; ++mt) {
            // Q A-fragment (K = hd = 32 -> single WMMA per S tile)
            Frag qa;
            const bf16_t* qp = qk + (mt * 16 + ln16) * QK_STRIDE + head * HD + g * 8;
            qa.h[0] = *(const v8bf*)(qp);
            qa.h[1] = *(const v8bf*)(qp + 16);
            v8f s[4];
            #pragma unroll
            for (int nt = 0; nt < 4; ++nt) {
                Frag kb;   // B-frag: col = key token, K = feature (token-major qk)
                const bf16_t* kp = qk + (nt * 16 + ln16) * QK_STRIDE + 256 + head * HD + g * 8;
                kb.h[0] = *(const v8bf*)(kp);
                kb.h[1] = *(const v8bf*)(kp + 16);
                v8f z = {};
                s[nt] = __builtin_amdgcn_wmma_f32_16x16x32_bf16(
                    false, qa.v, false, kb.v, (short)0, z, false, false);
            }
            // additive window mask
            #pragma unroll
            for (int nt = 0; nt < 4; ++nt)
                #pragma unroll
                for (int r = 0; r < 8; ++r)
                    s[nt][r] += mk[(mt * 16 + r + 8 * g) * 64 + nt * 16 + ln16];
            // row softmax: each row lives in one 16-lane half (C/D layout)
            float inv[8];
            #pragma unroll
            for (int r = 0; r < 8; ++r) {
                float mx = fmaxf(fmaxf(s[0][r], s[1][r]), fmaxf(s[2][r], s[3][r]));
                mx = fmaxf(mx, __shfl_xor(mx, 1, 32));
                mx = fmaxf(mx, __shfl_xor(mx, 2, 32));
                mx = fmaxf(mx, __shfl_xor(mx, 4, 32));
                mx = fmaxf(mx, __shfl_xor(mx, 8, 32));
                float sm = 0.0f;
                #pragma unroll
                for (int nt = 0; nt < 4; ++nt) {
                    s[nt][r] = __expf(s[nt][r] - mx);
                    sm += s[nt][r];
                }
                sm += __shfl_xor(sm, 1, 32);
                sm += __shfl_xor(sm, 2, 32);
                sm += __shfl_xor(sm, 4, 32);
                sm += __shfl_xor(sm, 8, 32);
                inv[r] = __builtin_amdgcn_rcpf(sm);
            }
            // P slab -> LDS (bf16, padded 16x64, private per wave)
            #pragma unroll
            for (int nt = 0; nt < 4; ++nt)
                #pragma unroll
                for (int r = 0; r < 8; ++r)
                    myP[(r + 8 * g) * PB_STRIDE + nt * 16 + ln16] =
                        (bf16_t)(s[nt][r] * inv[r]);
            // O slab = P @ V  (16x32, K = 64 -> 2 steps)
            #pragma unroll
            for (int ot = 0; ot < 2; ++ot) {
                v8f o = {};
                #pragma unroll
                for (int ks = 0; ks < 2; ++ks) {
                    Frag pa;
                    const bf16_t* pp = myP + ln16 * PB_STRIDE + ks * 32 + g * 8;
                    pa.h[0] = *(const v8bf*)(pp);
                    pa.h[1] = *(const v8bf*)(pp + 16);
                    Frag vb;   // B-frag from transposed V: col = feat, K = token
                    const bf16_t* vp = vT + (head * HD + ot * 16 + ln16) * VT_STRIDE
                                          + ks * 32 + g * 8;
                    vb.h[0] = *(const v8bf*)(vp);
                    vb.h[1] = *(const v8bf*)(vp + 16);
                    o = __builtin_amdgcn_wmma_f32_16x16x32_bf16(
                        false, pa.v, false, vb.v, (short)0, o, false, false);
                }
                #pragma unroll
                for (int r = 0; r < 8; ++r)
                    aout[(mt * 16 + r + 8 * g) * XB_STRIDE + head * HD + ot * 16 + ln16] =
                        (bf16_t)o[r];
            }
        }
    }
    __syncthreads();

    // ---------------- Phase 3: out = aout @ Wproj + b  (64x256 @ 256x256) -> global f32
    // Same structure: each wave owns 2 n-tiles across all 4 m-tiles.
    {
        float* og = out + (size_t)b * (NTOK * CDIM);
        for (int j = 0; j < 2; ++j) {
            const int nt = wv + 8 * j;
            v8f acc[4] = {{}, {}, {}, {}};
            const bf16_t* wp = wproj_pk + nt * (8 * 512) + lane * 16;
            const bf16_t* ap0 = aout + ln16 * XB_STRIDE + g * 8;
            #pragma unroll
            for (int kt = 0; kt < 8; ++kt) {
                Frag bb;
                bb.h[0] = *(const v8bf*)(wp + kt * 512);
                bb.h[1] = *(const v8bf*)(wp + kt * 512 + 8);
                #pragma unroll
                for (int mt = 0; mt < 4; ++mt) {
                    Frag a;
                    const bf16_t* ap = ap0 + mt * 16 * XB_STRIDE + kt * 32;
                    a.h[0] = *(const v8bf*)(ap);
                    a.h[1] = *(const v8bf*)(ap + 16);
                    acc[mt] = __builtin_amdgcn_wmma_f32_16x16x32_bf16(
                        false, a.v, false, bb.v, (short)0, acc[mt], false, false);
                }
            }
            const int col = nt * 16 + ln16;
            const float bias = b_proj[col];
            #pragma unroll
            for (int mt = 0; mt < 4; ++mt)
                #pragma unroll
                for (int r = 0; r < 8; ++r)
                    og[(mt * 16 + r + 8 * g) * CDIM + col] = acc[mt][r] + bias;
        }
    }
}

extern "C" void kernel_launch(void* const* d_in, const int* in_sizes, int n_in,
                              void* d_out, int out_size, void* d_ws, size_t ws_size,
                              hipStream_t stream) {
    const float* x      = (const float*)d_in[0];
    const float* mask   = (const float*)d_in[1];
    const float* w_qkv  = (const float*)d_in[2];
    const float* b_qkv  = (const float*)d_in[3];
    const float* w_proj = (const float*)d_in[4];
    const float* b_proj = (const float*)d_in[5];

    bf16_t* wqkv_pk  = (bf16_t*)d_ws;
    bf16_t* wproj_pk = (bf16_t*)((char*)d_ws + (size_t)WQKV_ELEMS * sizeof(bf16_t));

    pack_weights_kernel<<<(WQKV_ELEMS + 255) / 256, 256, 0, stream>>>(
        w_qkv, w_proj, wqkv_pk, wproj_pk);

    window_attn_kernel<<<1024, 256, LDS_BYTES, stream>>>(
        x, mask, b_qkv, b_proj, wqkv_pk, wproj_pk, (float*)d_out);
}
